// SequenceModel_7679401525513
// MI455X (gfx1250) — matrix operand order repaired and checked by reading, not compiled, not run on hardware
//
#include <hip/hip_runtime.h>
#include <stdint.h>

#define BATCH 64
#define TLEN  512
#define DIN   512
#define HDIM  1024
#define NWG   16    // persistent workgroups in recurrence

typedef __attribute__((ext_vector_type(16))) __bf16   v16bf;
typedef __attribute__((ext_vector_type(8)))  float    v8f;

struct U4x2 { uint4 a, b; };

// 16 contiguous bf16 (32B, 16B-aligned) -> v16bf
static __device__ inline v16bf bf16x16_load(const __bf16* p) {
    U4x2 t;
    t.a = *(const uint4*)p;
    t.b = *(const uint4*)(p + 8);
    return __builtin_bit_cast(v16bf, t);
}

// A-matrix (16x32, 16-bit): lane L<16 holds row L, K=0..15;
// lane L>=16 holds row L-16, K=16..31 (matches B). Source f32 row-major (ld floats).
static __device__ inline v16bf load_a_f32(const float* base, int ld) {
    const int lane = threadIdx.x & 31;
    const int r    = lane & 15;
    const int hi   = lane >> 4;
    const float* p = base + r * ld + hi * 16;
    v8f x0 = *(const v8f*)p;          // K run 0
    v8f x1 = *(const v8f*)(p + 8);    // K run 1 (+8)
    v16bf a;
#pragma unroll
    for (int e = 0; e < 8; ++e) {
        a[e]     = (__bf16)x0[e];
        a[e + 8] = (__bf16)x1[e];
    }
    return a;
}

// Same A layout, source already bf16 (recurrence h mirror): two 16B loads.
static __device__ inline v16bf load_a_bf16(const __bf16* base, int ld) {
    const int lane = threadIdx.x & 31;
    const int r    = lane & 15;
    const int hi   = lane >> 4;
    const __bf16* p = base + r * ld + hi * 16;
    U4x2 t;
    t.a = *(const uint4*)p;          // K=base..+7
    t.b = *(const uint4*)(p + 8);    // K=base+8..+15
    return __builtin_bit_cast(v16bf, t);
}

// B-matrix (32x16): column n of B = row n of row-major source (out = X @ W^T).
// lanes 0-15: col=lane, K=0..15 ; lanes 16-31: col=lane-16, K=16..31.
static __device__ inline v16bf load_b_rows(const __bf16* src, int ld) {
    const int lane = threadIdx.x & 31;
    const int n    = lane & 15;
    const int hi   = lane >> 4;
    return bf16x16_load(src + (size_t)n * ld + hi * 16);
}

static __device__ inline v8f wmma_bf16(v16bf a, v16bf b, v8f c) {
    return __builtin_amdgcn_wmma_f32_16x16x32_bf16(false, a, false, b,
                                                   (short)0, c, false, false);
}

// ---------------- weight f32 -> bf16 ----------------
__global__ void cvt_f32_bf16_kernel(const float* __restrict__ src,
                                    __bf16* __restrict__ dst, int n) {
    int i = blockIdx.x * blockDim.x + threadIdx.x;
    if (i < n) dst[i] = (__bf16)src[i];
}

// ---------------- phase 1: z[t][b][n] = tanh(X @ Wxh^T + bxh) + bhh ----------------
// grid: (BT/128, HDIM/64), block 256 (8 waves). Wave w -> rows [w*16,+16), 64 cols.
// z is stored time-major so each recurrence step touches one contiguous 128KB slab.
__global__ void xproj_kernel(const float* __restrict__ x,
                             const __bf16* __restrict__ wxh,
                             const float* __restrict__ bxh,
                             const float* __restrict__ bhh,
                             __bf16* __restrict__ z) {
    const int wave = threadIdx.x >> 5;
    const int lane = threadIdx.x & 31;
    const int m0 = blockIdx.x * 128 + wave * 16;
    const int n0 = blockIdx.y * 64;

    v8f acc[4] = {};
    for (int k0 = 0; k0 < DIN; k0 += 32) {
        v16bf a = load_a_f32(x + (size_t)m0 * DIN + k0, DIN);
#pragma unroll
        for (int j = 0; j < 4; ++j) {
            v16bf b = load_b_rows(wxh + (size_t)(n0 + 16 * j) * DIN + k0, DIN);
            acc[j] = wmma_bf16(a, b, acc[j]);
        }
    }
    const int hi = lane >> 4;
    const int nl = lane & 15;
#pragma unroll
    for (int j = 0; j < 4; ++j) {
        const int col = n0 + 16 * j + nl;
        const float bias  = bxh[col];
        const float biash = bhh[col];
#pragma unroll
        for (int i = 0; i < 8; ++i) {
            const int row = m0 + hi * 8 + i;      // row m = b*TLEN + t
            const int b   = row >> 9;             // /TLEN
            const int t   = row & (TLEN - 1);
            float v = tanhf(acc[j][i] + bias) + biash;
            z[((size_t)t * BATCH + b) * HDIM + col] = (__bf16)v;
        }
    }
}

// ---------------- phase 2: persistent recurrence ----------------
// 16 WGs, WG g owns output cols [g*64, g*64+64). W_hh tile pinned in LDS
// (rows padded to 1032 halfs -> conflict-free ds_load_b128).
// f32 carry lives in REGISTERS (each wave re-reads exactly what it wrote);
// only the bf16 mirror (WMMA A operand for all WGs) ping-pongs through L2.
__global__ void recur_kernel(const __bf16* __restrict__ z,
                             const __bf16* __restrict__ whh,
                             const int* __restrict__ seqlen,
                             __bf16* __restrict__ hbf0,
                             __bf16* __restrict__ hbf1,
                             float* __restrict__ hfin,
                             unsigned int* __restrict__ counter) {
    extern __shared__ __align__(16) char smem[];
    __bf16* Wl = (__bf16*)smem;                 // [64][1032] halfs = 132 KB
    __shared__ int slen[BATCH];

    const int g    = blockIdx.x;
    const int tid  = threadIdx.x;
    const int wave = tid >> 5;
    const int lane = tid & 31;
    const int nl   = lane & 15;
    const int hi   = lane >> 4;

    // preload this WG's 64 rows of W_hh (each row = one output column), 16B chunks
    for (int idx = tid; idx < 64 * 128; idx += 256) {
        const int r = idx >> 7, c = idx & 127;
        *(uint4*)(Wl + r * 1032 + c * 8) =
            *(const uint4*)(whh + ((size_t)(g * 64 + r)) * HDIM + c * 8);
    }
    if (tid < BATCH) slen[tid] = seqlen[tid];
    __syncthreads();

    const int mi  = wave & 3;              // M tile (rows mi*16..+16)
    const int ni0 = (wave >> 2) * 2;       // two N tiles per wave
    const int ni1 = ni0 + 1;
    const int col0 = g * 64 + ni0 * 16 + nl;
    const int col1 = g * 64 + ni1 * 16 + nl;

    __bf16* curbf = hbf0;
    __bf16* nxtbf = hbf1;

    float hreg0[8] = {}, hreg1[8] = {};    // f32 carry, register-resident

    for (int t = 0; t < TLEN; ++t) {
        v8f acc0 = {}, acc1 = {};
        for (int k0 = 0; k0 < HDIM; k0 += 32) {
            v16bf a  = load_a_bf16(curbf + (size_t)(mi * 16) * HDIM + k0, HDIM);
            v16bf b0 = bf16x16_load(Wl + (ni0 * 16 + nl) * 1032 + k0 + hi * 16);
            v16bf b1 = bf16x16_load(Wl + (ni1 * 16 + nl) * 1032 + k0 + hi * 16);
            acc0 = wmma_bf16(a, b0, acc0);
            acc1 = wmma_bf16(a, b1, acc1);
        }
        const __bf16* zt = z + (size_t)t * BATCH * HDIM;
#pragma unroll
        for (int i = 0; i < 8; ++i) {
            const int m = mi * 16 + hi * 8 + i;          // batch row
            const bool active = (t < slen[m]);
            const float v0 = acc0[i] + (float)zt[(size_t)m * HDIM + col0];
            const float v1 = acc1[i] + (float)zt[(size_t)m * HDIM + col1];
            hreg0[i] = active ? v0 : hreg0[i];
            hreg1[i] = active ? v1 : hreg1[i];
            nxtbf[(size_t)m * HDIM + col0] = (__bf16)hreg0[i];
            nxtbf[(size_t)m * HDIM + col1] = (__bf16)hreg1[i];
        }
        // device-wide barrier (monotonic counter, reset per launch)
        __threadfence();
        __syncthreads();
        if (tid == 0) {
            __hip_atomic_fetch_add(counter, 1u, __ATOMIC_ACQ_REL,
                                   __HIP_MEMORY_SCOPE_AGENT);
            const unsigned int target = (unsigned)NWG * (unsigned)(t + 1);
            while (__hip_atomic_load(counter, __ATOMIC_ACQUIRE,
                                     __HIP_MEMORY_SCOPE_AGENT) < target) {
                __builtin_amdgcn_s_sleep(8);
            }
        }
        __syncthreads();
        __threadfence();
        __bf16* tb = curbf; curbf = nxtbf; nxtbf = tb;
    }

    // dump exact f32 carry for the output projection
#pragma unroll
    for (int i = 0; i < 8; ++i) {
        const int m = mi * 16 + hi * 8 + i;
        hfin[(size_t)m * HDIM + col0] = hreg0[i];
        hfin[(size_t)m * HDIM + col1] = hreg1[i];
    }
}

// ---------------- phase 3: out = hT @ W_out^T + b_out ----------------
__global__ void outproj_kernel(const float* __restrict__ h,
                               const float* __restrict__ wout,
                               const float* __restrict__ bout,
                               float* __restrict__ out) {
    const int b = threadIdx.x;
    if (b < BATCH) {
        float acc = bout[0];
        for (int k = 0; k < HDIM; ++k)
            acc += h[(size_t)b * HDIM + k] * wout[k];
        out[b] = acc;
    }
}

extern "C" void kernel_launch(void* const* d_in, const int* in_sizes, int n_in,
                              void* d_out, int out_size, void* d_ws, size_t ws_size,
                              hipStream_t stream) {
    (void)in_sizes; (void)n_in; (void)out_size; (void)ws_size;
    const float* x    = (const float*)d_in[0];
    const int*   sl   = (const int*)  d_in[1];
    const float* wxh  = (const float*)d_in[2];
    const float* bxh  = (const float*)d_in[3];
    const float* whh  = (const float*)d_in[4];
    const float* bhh  = (const float*)d_in[5];
    const float* wout = (const float*)d_in[6];
    const float* bout = (const float*)d_in[7];
    float* out = (float*)d_out;

    char* ws = (char*)d_ws;
    size_t o = 0;
    __bf16* wxh_bf = (__bf16*)(ws + o); o += (size_t)HDIM * DIN * 2;
    __bf16* whh_bf = (__bf16*)(ws + o); o += (size_t)HDIM * HDIM * 2;
    __bf16* z      = (__bf16*)(ws + o); o += (size_t)BATCH * TLEN * HDIM * 2;
    __bf16* hb0    = (__bf16*)(ws + o); o += (size_t)BATCH * HDIM * 2;
    __bf16* hb1    = (__bf16*)(ws + o); o += (size_t)BATCH * HDIM * 2;
    float*  hfin   = (float*) (ws + o); o += (size_t)BATCH * HDIM * 4;
    unsigned int* counter = (unsigned int*)(ws + o);

    // deterministic per-call init (workspace is poisoned, not re-poisoned)
    hipMemsetAsync(hb0, 0, (size_t)BATCH * HDIM * 2, stream);   // h(-1) = 0
    hipMemsetAsync(counter, 0, sizeof(unsigned int), stream);

    cvt_f32_bf16_kernel<<<(HDIM * DIN  + 255) / 256, 256, 0, stream>>>(wxh, wxh_bf, HDIM * DIN);
    cvt_f32_bf16_kernel<<<(HDIM * HDIM + 255) / 256, 256, 0, stream>>>(whh, whh_bf, HDIM * HDIM);

    dim3 g1(BATCH * TLEN / 128, HDIM / 64);
    xproj_kernel<<<g1, 256, 0, stream>>>(x, wxh_bf, bxh, bhh, z);

    const size_t smem = (size_t)64 * 1032 * sizeof(__bf16);   // 132 KB of 320 KB LDS
    recur_kernel<<<NWG, 256, smem, stream>>>(z, whh_bf, sl, hb0, hb1, hfin, counter);

    outproj_kernel<<<1, 64, 0, stream>>>(hfin, wout, bout, out);
}
